// ScaledDotProductAttentionModel_4509715661508
// MI455X (gfx1250) — matrix operand order repaired and checked by reading, not compiled, not run on hardware
//
#include <hip/hip_runtime.h>
#include <hip/hip_bf16.h>

#define EMB 1024
#define HEADS 16
#define HEAD_DIM 64
#define NTOK 16384

typedef __attribute__((ext_vector_type(16))) __bf16 v16bf;
typedef __attribute__((ext_vector_type(8)))  float  v8f;

union FragAB {
  v16bf v;
  uint4 u4[2];
  unsigned u32[8];
};
union FragC {
  v8f v;
  float f[8];
};

__device__ __forceinline__ unsigned short f2bf(float x) {
  unsigned u = __float_as_uint(x);
  unsigned r = u + 0x7FFFu + ((u >> 16) & 1u);   // round-to-nearest-even
  return (unsigned short)(r >> 16);
}
__device__ __forceinline__ unsigned pack2bf(float lo, float hi) {
  return (unsigned)f2bf(lo) | ((unsigned)f2bf(hi) << 16);
}

// single-instruction lane-XOR exchange within 16-lane groups (ds_swizzle_b32)
#define SWZ_XOR(x, m) \
  __uint_as_float((unsigned)__builtin_amdgcn_ds_swizzle( \
      (int)__float_as_uint(x), (((m) << 10) | 0x1f)))

// ---------------------------------------------------------------------------
// Async global->LDS copy (ASYNCcnt path). The builtin takes int4 pointers:
// (v4i addrspace(1)*, v4i addrspace(3)*, imm offset, imm cpol).
// Falls back to load+ds_store if the builtin is absent.
// ---------------------------------------------------------------------------
#if __has_builtin(__builtin_amdgcn_global_load_async_to_lds_b128)
#define HAVE_ASYNC_LDS 1
#else
#define HAVE_ASYNC_LDS 0
#endif

typedef int v4i __attribute__((ext_vector_type(4)));
typedef __attribute__((address_space(1))) v4i gv4i;  // global int4
typedef __attribute__((address_space(3))) v4i sv4i;  // LDS int4

__device__ __forceinline__ void copy16_g2l(const void* g, void* l) {
#if HAVE_ASYNC_LDS
  __builtin_amdgcn_global_load_async_to_lds_b128((gv4i*)g, (sv4i*)l, 0, 0);
#else
  *(uint4*)l = *(const uint4*)g;
#endif
}
__device__ __forceinline__ void wait_async_all() {
#if HAVE_ASYNC_LDS
  asm volatile("s_wait_asynccnt 0" ::: "memory");
#endif
}

// ---------------------------------------------------------------------------
// fp32 -> bf16 conversion, 2 elements per thread, packed dword store
// ---------------------------------------------------------------------------
__global__ __launch_bounds__(256) void cvt_kernel(const float* __restrict__ in,
                                                  unsigned* __restrict__ out, int n2) {
  int i = blockIdx.x * blockDim.x + threadIdx.x;
  if (i < n2) out[i] = pack2bf(in[2 * i], in[2 * i + 1]);
}

// ---------------------------------------------------------------------------
// GEMM: Y[M,N] = A[M,K] @ W[N,K]^T + bias, bf16 WMMA, fp32 accumulate.
// Block = 256 threads (8 waves), block tile 128x256, wave tile 64x64.
// Double-buffered LDS pipeline: async global->LDS copies of the next K-step
// (A 128x32 = 8KB, B 256x32 = 16KB) overlap with 128 WMMAs on the current
// K-step, fragments gathered from LDS with ds_load_b128.
// ---------------------------------------------------------------------------
#define A_TILE_BYTES 8192    // 128 rows * 64B
#define B_TILE_BYTES 16384   // 256 rows * 64B
#define BUF_BYTES (A_TILE_BYTES + B_TILE_BYTES)

template <bool OUT_F32>
__global__ __launch_bounds__(256) void gemm_wmma(const unsigned short* __restrict__ A,
                                                 const unsigned short* __restrict__ W,
                                                 const float* __restrict__ bias,
                                                 void* __restrict__ Out,
                                                 int M, int K, int N) {
  __shared__ unsigned char lds[2 * BUF_BYTES];  // 48KB: {A,B} x double buffer

  const int tid  = threadIdx.x;
  const int lane = tid & 31;
  const int wid  = tid >> 5;
  const int half = lane >> 4;
  const int l16  = lane & 15;
  const int mBlock = blockIdx.y * 128;
  const int nBlock = blockIdx.x * 256;
  const int mWave = (wid >> 2) * 64;  // within block tile
  const int nWave = (wid & 3) * 64;

  // per-thread copy assignments (16B chunks): A has 512 chunks, B has 1024
  auto issue = [&](int buf, int kb) {
    unsigned char* Al = lds + buf * BUF_BYTES;
    unsigned char* Bl = Al + A_TILE_BYTES;
    for (int i = 0; i < 2; ++i) {             // A tile: 2 chunks/thread
      const int chunk = i * 256 + tid;
      const int row = chunk >> 2, c4 = chunk & 3;
      copy16_g2l(A + (size_t)(mBlock + row) * K + kb + c4 * 8, Al + row * 64 + c4 * 16);
    }
    for (int i = 0; i < 4; ++i) {             // B tile: 4 chunks/thread
      const int chunk = i * 256 + tid;
      const int row = chunk >> 2, c4 = chunk & 3;
      copy16_g2l(W + (size_t)(nBlock + row) * K + kb + c4 * 8, Bl + row * 64 + c4 * 16);
    }
  };

  FragC c[4][4];
  for (int i = 0; i < 4; ++i)
    for (int j = 0; j < 4; ++j)
      for (int r = 0; r < 8; ++r) c[i][j].f[r] = 0.f;

  const int nk = K / 32;
  issue(0, 0);
  for (int i = 0; i < nk; ++i) {
    const int cur = i & 1;
    wait_async_all();     // this wave's copies into buffer `cur` are in LDS
    __syncthreads();      // all waves' copies visible; all done reading cur^1
    if (i + 1 < nk) issue(cur ^ 1, (i + 1) * 32);  // overlaps with compute below

    const unsigned char* Al = lds + cur * BUF_BYTES;
    const unsigned char* Bl = Al + A_TILE_BYTES;

    FragAB a[4];
    for (int ms = 0; ms < 4; ++ms) {
      const unsigned char* ap = Al + (mWave + ms * 16 + l16) * 64 + half * 16;
      a[ms].u4[0] = *(const uint4*)ap;
      a[ms].u4[1] = *(const uint4*)(ap + 32);
    }
    for (int ns = 0; ns < 4; ++ns) {
      FragAB b;
      const unsigned char* bp = Bl + (nWave + ns * 16 + l16) * 64 + half * 16;
      b.u4[0] = ((const uint4*)bp)[0];
      b.u4[1] = ((const uint4*)bp)[2];
      for (int ms = 0; ms < 4; ++ms)
        c[ms][ns].v = __builtin_amdgcn_wmma_f32_16x16x32_bf16(
            false, a[ms].v, false, b.v, (short)0, c[ms][ns].v, false, false);
    }
    __syncthreads();  // done reading buffer cur before it is refilled at i+2
  }

  for (int ns = 0; ns < 4; ++ns) {
    const int n = nBlock + nWave + ns * 16 + l16;
    const float bv = bias[n];
    for (int ms = 0; ms < 4; ++ms)
      for (int r = 0; r < 8; ++r) {
        const int m = mBlock + mWave + ms * 16 + half * 8 + r;
        const float val = c[ms][ns].f[r] + bv;
        if (OUT_F32)
          ((float*)Out)[(size_t)m * N + n] = val;
        else
          ((unsigned short*)Out)[(size_t)m * N + n] = f2bf(val);
      }
  }
}

// ---------------------------------------------------------------------------
// Per-token attention across heads: scores[n] = q[n] @ k[n]^T (16x16, K=64),
// softmax over last axis, ctx[n] = attn @ v[n] (16x64, K=16 padded to 32).
// One wave per token, 8 tokens per block.
// ---------------------------------------------------------------------------
__global__ __launch_bounds__(256) void attn_kernel(const unsigned short* Q,
                                                   const unsigned short* Kd,
                                                   const unsigned short* Vd,
                                                   unsigned short* Ctx) {
  __shared__ unsigned short smem[8 * 1280];  // per wave: vT[64][16] + attn[16][16]
  const int lane = threadIdx.x & 31;
  const int wid  = threadIdx.x >> 5;
  const int half = lane >> 4;
  const int l16  = lane & 15;
  const int tok  = blockIdx.x * 8 + wid;

  unsigned short* vT = &smem[wid * 1280];
  unsigned short* at = vT + 1024;

  // stage vT[d][g] = v[g][d]  (makes B columns of attn@v contiguous over K=g)
  const unsigned short* vsrc = Vd + (size_t)tok * EMB;
  for (int j = 0; j < 8; ++j) {
    int e = lane * 32 + j * 4;       // 4 consecutive d within one head row g
    int g = e >> 6, d = e & 63;
    uint2 x = *(const uint2*)(vsrc + e);
    vT[(d + 0) * 16 + g] = (unsigned short)(x.x & 0xffff);
    vT[(d + 1) * 16 + g] = (unsigned short)(x.x >> 16);
    vT[(d + 2) * 16 + g] = (unsigned short)(x.y & 0xffff);
    vT[(d + 3) * 16 + g] = (unsigned short)(x.y >> 16);
  }

  // scores = q @ k^T : two 16x16x32 bf16 WMMAs over D=64
  FragC s;
  for (int r = 0; r < 8; ++r) s.f[r] = 0.f;
  const unsigned short* qsrc = Q + (size_t)tok * EMB;
  const unsigned short* ksrc = Kd + (size_t)tok * EMB;
  for (int kb = 0; kb < 64; kb += 32) {
    FragAB a, b;
    a.u4[0] = *(const uint4*)(qsrc + l16 * 64 + kb + half * 16);
    a.u4[1] = *(const uint4*)(qsrc + l16 * 64 + kb + half * 16 + 8);
    const uint4* bp = (const uint4*)(ksrc + l16 * 64 + kb + half * 16);
    b.u4[0] = bp[0];
    b.u4[1] = bp[1];
    s.v = __builtin_amdgcn_wmma_f32_16x16x32_bf16(
        false, a.v, false, b.v, (short)0, s.v, false, false);
  }

  // row-wise softmax across the 16 lanes of each half (ds_swizzle XOR)
  float p[8];
  for (int r = 0; r < 8; ++r) {
    float x = s.f[r] * 0.125f;  // 1/sqrt(64)
    float mx = x;
    mx = fmaxf(mx, SWZ_XOR(mx, 1));
    mx = fmaxf(mx, SWZ_XOR(mx, 2));
    mx = fmaxf(mx, SWZ_XOR(mx, 4));
    mx = fmaxf(mx, SWZ_XOR(mx, 8));
    float e = __expf(x - mx);
    float sum = e;
    sum += SWZ_XOR(sum, 1);
    sum += SWZ_XOR(sum, 2);
    sum += SWZ_XOR(sum, 4);
    sum += SWZ_XOR(sum, 8);
    p[r] = e / sum;
  }

  // stage attn (bf16) in LDS for re-layout C->A
  for (int r = 0; r < 8; ++r) {
    int m = r + half * 8;
    at[m * 16 + l16] = f2bf(p[r]);
  }
  __syncthreads();  // covers vT and at visibility

  // A = attn (K=16 padded to 32: contiguous k, K=16..31 meets B's zeros)
  FragAB a2;
  a2.u4[0] = *(const uint4*)(at + l16 * 16);
  a2.u4[1] = *(const uint4*)(at + l16 * 16 + 8);

  unsigned short* cdst = Ctx + (size_t)tok * EMB;
  for (int t = 0; t < 4; ++t) {
    FragAB b2;
    if (half == 0) {  // K=0..15 real; K=16..31 is B's zeros (A repeats attn there)
      const uint4* bp = (const uint4*)(vT + (t * 16 + l16) * 16);
      b2.u4[0] = bp[0];
      b2.u4[1] = bp[1];
    } else {
      b2.u4[0] = make_uint4(0, 0, 0, 0);
      b2.u4[1] = make_uint4(0, 0, 0, 0);
    }
    FragC o;
    for (int r = 0; r < 8; ++r) o.f[r] = 0.f;
    o.v = __builtin_amdgcn_wmma_f32_16x16x32_bf16(
        false, a2.v, false, b2.v, (short)0, o.v, false, false);
    for (int r = 0; r < 8; ++r) {
      int h = r + half * 8;
      int d = t * 16 + l16;
      cdst[h * 64 + d] = f2bf(o.f[r]);
    }
  }
}

// ---------------------------------------------------------------------------
extern "C" void kernel_launch(void* const* d_in, const int* in_sizes, int n_in,
                              void* d_out, int out_size, void* d_ws, size_t ws_size,
                              hipStream_t stream) {
  const float* src = (const float*)d_in[0];
  const float* Wq  = (const float*)d_in[1];
  const float* bq  = (const float*)d_in[2];
  const float* Wk  = (const float*)d_in[3];
  const float* bk  = (const float*)d_in[4];
  const float* Wv  = (const float*)d_in[5];
  const float* bv  = (const float*)d_in[6];
  const float* Wo  = (const float*)d_in[7];
  const float* bo  = (const float*)d_in[8];
  float* out = (float*)d_out;

  const size_t WSZ = (size_t)EMB * EMB;   // 1M elems per weight
  const size_t QSZ = (size_t)NTOK * EMB;  // 16M elems per activation
  unsigned short* ws  = (unsigned short*)d_ws;
  unsigned short* Wqb = ws;
  unsigned short* Wkb = Wqb + WSZ;
  unsigned short* Wvb = Wkb + WSZ;
  unsigned short* Wob = Wvb + WSZ;
  unsigned short* Xb  = Wob + WSZ;   // src in bf16
  unsigned short* Qb  = Xb + QSZ;
  unsigned short* Kb  = Qb + QSZ;
  unsigned short* Vb  = Kb + QSZ;
  unsigned short* Ctx = Qb;  // in-place: each wave finishes q reads before ctx writes

  {
    int n2 = (int)(WSZ / 2);
    int blocks = (n2 + 255) / 256;
    cvt_kernel<<<blocks, 256, 0, stream>>>(Wq, (unsigned*)Wqb, n2);
    cvt_kernel<<<blocks, 256, 0, stream>>>(Wk, (unsigned*)Wkb, n2);
    cvt_kernel<<<blocks, 256, 0, stream>>>(Wv, (unsigned*)Wvb, n2);
    cvt_kernel<<<blocks, 256, 0, stream>>>(Wo, (unsigned*)Wob, n2);
    int n2x = (int)(QSZ / 2);
    cvt_kernel<<<(n2x + 255) / 256, 256, 0, stream>>>(src, (unsigned*)Xb, n2x);
  }

  dim3 gg(EMB / 256, NTOK / 128);
  gemm_wmma<false><<<gg, 256, 0, stream>>>(Xb, Wqb, bq, Qb, NTOK, EMB, EMB);
  gemm_wmma<false><<<gg, 256, 0, stream>>>(Xb, Wkb, bk, Kb, NTOK, EMB, EMB);
  gemm_wmma<false><<<gg, 256, 0, stream>>>(Xb, Wvb, bv, Vb, NTOK, EMB, EMB);

  attn_kernel<<<NTOK / 8, 256, 0, stream>>>(Qb, Kb, Vb, Ctx);

  gemm_wmma<true><<<gg, 256, 0, stream>>>(Ctx, Wob, bo, out, NTOK, EMB, EMB);
}
